// CombinedAttention_84516366450970
// MI455X (gfx1250) — matrix operand order repaired and checked by reading, not compiled, not run on hardware
//
#include <hip/hip_runtime.h>
#include <hip/hip_bf16.h>

// ---------------------------------------------------------------------------
// CombinedAttention for MI455X (gfx1250): all GEMMs on v_wmma_f32_16x16x32_f16,
// flash-style fused softmax(QsKs^T + QcKc^T) @ V, f32 accumulation throughout.
// Round 2: LDS-shared K/V tiles in attention (8x traffic cut), 64-key tiles
// (half the softmax shuffle/correction work per key), 64-deep K-slabs in the
// projection GEMMs (half the barrier count).
// ---------------------------------------------------------------------------

typedef _Float16 f16;
typedef f16   v8h  __attribute__((ext_vector_type(8)));
typedef f16   v16h __attribute__((ext_vector_type(16)));
typedef float v8f  __attribute__((ext_vector_type(8)));

#define HEADS   8
#define NSEQ    2048
#define DHEAD   32
#define LATENT  256
#define ATTN_SCALE 0.17677669529663687f  // 32^-0.5

__device__ __forceinline__ v8f wmma_f16(v16h a, v16h b, v8f c) {
  return __builtin_amdgcn_wmma_f32_16x16x32_f16(
      /*neg_a=*/false, a, /*neg_b=*/false, b,
      /*c_mod=*/(short)0, c, /*reuse_a=*/false, /*reuse_b=*/false);
}

__device__ __forceinline__ v16h frag_from2(const f16* lo, const f16* hi) {
  v8h a = *(const v8h*)lo;
  v8h b = *(const v8h*)hi;
  return __builtin_shufflevector(a, b, 0,1,2,3,4,5,6,7,8,9,10,11,12,13,14,15);
}

// A-fragment (16x32 MxK), ISA layout: lane<16 -> K 0..7 & 16..23 of row lane;
// lane>=16 -> K 8..15 & 24..31 of row lane-16.
__device__ __forceinline__ v16h load_a_frag(const f16* p, int row0, int stride, int lane) {
  int m    = lane & 15;
  int koff = (lane >> 4) << 3;
  const f16* r = p + (size_t)(row0 + m) * stride + koff;
  return frag_from2(r, r + 16);
}

// B-fragment (32x16 KxN), K along rows of the buffer: lane l holds column
// n0 + (l&15), K 0..7 & 16..23 (lane<16) or K 8..15 & 24..31 (lane>=16).
__device__ __forceinline__ v16h load_b_frag(const f16* p, int stride, int n0, int lane) {
  const int koff = (lane >> 4) << 3;
  const f16* r = p + (size_t)koff * stride + n0 + (lane & 15);
  v16h v;
#pragma unroll
  for (int s = 0; s < 8; ++s) {
    v[s]     = r[(size_t)s * stride];
    v[s + 8] = r[(size_t)(s + 16) * stride];
  }
  return v;
}

// ---------------------------------------------------------------------------
__global__ void cvt_f32_to_f16_kernel(const float* __restrict__ x,
                                      f16* __restrict__ y, int n) {
  int i = blockIdx.x * blockDim.x + threadIdx.x;
  if (i < n) y[i] = (f16)x[i];
}

// ---------------------------------------------------------------------------
// Tiled GEMM: C[M,N] = X[M,K](f16) @ W[K,N](f16) + bias(f32)
//   MODE 0: write f16 split-head      [B,H,N,32]  (Q_self, Q_cross, V)
//   MODE 1: write f16 head-transposed [B,H,32,N]  (K_self, K_cross)
//   MODE 2: write f32 plain [M,N]                 (final output projection)
// Block tile 64x64, 4 waves (128 thr), wave tile 32x32, 64-deep K-slab per
// barrier pair (8 WMMAs between syncs).
// ---------------------------------------------------------------------------
template <int MODE>
__global__ void gemm_f16_kernel(const f16* __restrict__ X,
                                const f16* __restrict__ W,
                                const float* __restrict__ bias,
                                void* __restrict__ out,
                                int M, int K, int N) {
  __shared__ f16 Xs[64 * 72];  // 64 rows x 64 K, stride 72 (16B-aligned rows)
  __shared__ f16 Ws[64 * 72];  // 64 K   x 64 N, stride 72

  const int tid  = threadIdx.x;
  const int lane = tid & 31;
  const int wave = tid >> 5;
  const int m0 = blockIdx.y * 64;
  const int n0 = blockIdx.x * 64;
  const int wm = (wave >> 1) * 32;
  const int wn = (wave & 1) * 32;

  v8f acc[2][2] = {};

  const int srow = tid >> 1;          // 0..63
  const int soff = (tid & 1) * 32;    // 0 or 32 (each thread: 32 halfs/tile)

  for (int k0 = 0; k0 < K; k0 += 64) {
    // Stage 64x64 X-slab and 64x64 W-slab
    const f16* xsrc = X + (size_t)(m0 + srow) * K + k0 + soff;
    const f16* wsrc = W + (size_t)(k0 + srow) * N + n0 + soff;
#pragma unroll
    for (int q = 0; q < 4; ++q) {
      *(v8h*)&Xs[srow * 72 + soff + q * 8] = *(const v8h*)(xsrc + q * 8);
      *(v8h*)&Ws[srow * 72 + soff + q * 8] = *(const v8h*)(wsrc + q * 8);
    }
    if (k0 + 64 < K) {  // prefetch next K-slab (global_prefetch_b8)
      __builtin_prefetch(xsrc + 64, 0, 0);
      __builtin_prefetch(wsrc + (size_t)64 * N, 0, 0);
    }
    __syncthreads();

#pragma unroll
    for (int kk = 0; kk < 64; kk += 32) {
      v16h a0 = load_a_frag(Xs + kk, wm + 0,  72, lane);
      v16h a1 = load_a_frag(Xs + kk, wm + 16, 72, lane);
      v16h b0 = load_b_frag(Ws + (size_t)kk * 72, 72, wn + 0,  lane);
      v16h b1 = load_b_frag(Ws + (size_t)kk * 72, 72, wn + 16, lane);
      acc[0][0] = wmma_f16(a0, b0, acc[0][0]);
      acc[0][1] = wmma_f16(a0, b1, acc[0][1]);
      acc[1][0] = wmma_f16(a1, b0, acc[1][0]);
      acc[1][1] = wmma_f16(a1, b1, acc[1][1]);
    }
    __syncthreads();
  }

  // Epilogue. C layout: VGPR v -> row v (lanes 0-15) / v+8 (lanes 16-31),
  // col = lane&15.
  const int hi = lane >> 4;
  const int cl = lane & 15;
#pragma unroll
  for (int i = 0; i < 2; ++i) {
#pragma unroll
    for (int j = 0; j < 2; ++j) {
      const int cg = n0 + wn + j * 16 + cl;
      const float bv = bias[cg];
#pragma unroll
      for (int v = 0; v < 8; ++v) {
        const int rg = m0 + wm + i * 16 + v + hi * 8;
        const float val = acc[i][j][v] + bv;
        if (MODE == 2) {
          ((float*)out)[(size_t)rg * N + cg] = val;
        } else {
          const int b = rg >> 11, n = rg & (NSEQ - 1);
          const int h = cg >> 5,  d = cg & (DHEAD - 1);
          f16* o = (f16*)out;
          if (MODE == 0)
            o[(((size_t)(b * HEADS + h) * NSEQ) + n) * DHEAD + d] = (f16)val;
          else
            o[(((size_t)(b * HEADS + h) * DHEAD) + d) * NSEQ + n] = (f16)val;
        }
      }
    }
  }
}

// ---------------------------------------------------------------------------
// Flash attention: softmax( (Qs Ks^T + Qc Kc^T) * scale ) @ V
//   qs, qc : [B,H,N,32] f16      kst, kct : [B,H,32,N] f16 (pre-transposed)
//   vv     : [B,H,N,32] f16      oatt     : [B,N,256]  f16
// Grid: (N/128, B*H). 8 waves x 16 query rows; 64-key tiles staged in LDS
// once per block (shared by all 8 waves).
// ---------------------------------------------------------------------------
__global__ void attn_flash_kernel(const f16* __restrict__ qs,
                                  const f16* __restrict__ kst,
                                  const f16* __restrict__ qc,
                                  const f16* __restrict__ kct,
                                  const f16* __restrict__ vv,
                                  f16* __restrict__ oatt) {
  // Shared K/V tiles for the whole block + per-wave P staging
  __shared__ f16 Kss[32 * 72];      // Ks^T tile: 32 d-rows x 64 keys
  __shared__ f16 Kcs[32 * 72];      // Kc^T tile
  __shared__ f16 Vs [64 * 40];      // V tile: 64 key-rows x 32 d
  __shared__ f16 Ps [8][16 * 72];   // per-wave P: 16 rows x 64 keys

  const int tid  = threadIdx.x;
  const int lane = tid & 31;
  const int wave = tid >> 5;
  const int bh = blockIdx.y;
  const int b = bh >> 3, h = bh & (HEADS - 1);
  const int m0 = blockIdx.x * 128 + wave * 16;
  f16* myP = Ps[wave];

  // Q fragments (persist across the whole key loop)
  const v16h aQs = load_a_frag(qs + (size_t)bh * NSEQ * DHEAD, m0, DHEAD, lane);
  const v16h aQc = load_a_frag(qc + (size_t)bh * NSEQ * DHEAD, m0, DHEAD, lane);

  // Staging assignments (256 threads, 8 halfs each per 2048-half tile)
  const int krow = tid >> 3, kcol = (tid & 7) * 8;   // K tiles: 32 x 64
  const int vrow = tid >> 2, vcol = (tid & 3) * 8;   // V tile : 64 x 32
  const f16* ksg = kst + ((size_t)bh * DHEAD + krow) * NSEQ + kcol;
  const f16* kcg = kct + ((size_t)bh * DHEAD + krow) * NSEQ + kcol;
  const f16* vg  = vv  + ((size_t)bh * NSEQ + vrow) * DHEAD + vcol;

  float rmax[8], rsum[8];
#pragma unroll
  for (int v = 0; v < 8; ++v) { rmax[v] = -3.0e38f; rsum[v] = 0.0f; }
  v8f o0 = {}, o1 = {};

  const int hi = lane >> 4;
  const int cl = lane & 15;

  for (int j0 = 0; j0 < NSEQ; j0 += 64) {
    // Stage shared K/V tiles (WAR on previous iteration's reads)
    __syncthreads();
    *(v8h*)&Kss[krow * 72 + kcol] = *(const v8h*)(ksg + j0);
    *(v8h*)&Kcs[krow * 72 + kcol] = *(const v8h*)(kcg + j0);
    *(v8h*)&Vs [vrow * 40 + vcol] = *(const v8h*)(vg + (size_t)j0 * DHEAD);
    __syncthreads();

    // S[16 x 64] = Qc*Kc^T + Qs*Ks^T fused in the WMMA accumulator chain
    v8f s[4];
#pragma unroll
    for (int t = 0; t < 4; ++t) {
      v16h bKc = load_b_frag(Kcs, 72, t * 16, lane);
      v16h bKs = load_b_frag(Kss, 72, t * 16, lane);
      v8f z = {};
      z = wmma_f16(aQc, bKc, z);
      s[t] = wmma_f16(aQs, bKs, z);
    }

    // Online softmax over the 64-key tile
#pragma unroll
    for (int v = 0; v < 8; ++v) {
      float x0 = s[0][v] * ATTN_SCALE;
      float x1 = s[1][v] * ATTN_SCALE;
      float x2 = s[2][v] * ATTN_SCALE;
      float x3 = s[3][v] * ATTN_SCALE;
      float mx = fmaxf(fmaxf(x0, x1), fmaxf(x2, x3));
      mx = fmaxf(mx, __shfl_xor(mx, 1, 16));
      mx = fmaxf(mx, __shfl_xor(mx, 2, 16));
      mx = fmaxf(mx, __shfl_xor(mx, 4, 16));
      mx = fmaxf(mx, __shfl_xor(mx, 8, 16));
      const float mnew = fmaxf(rmax[v], mx);
      const float corr = __expf(rmax[v] - mnew);
      const float p0 = __expf(x0 - mnew);
      const float p1 = __expf(x1 - mnew);
      const float p2 = __expf(x2 - mnew);
      const float p3 = __expf(x3 - mnew);
      float ps = (p0 + p1) + (p2 + p3);
      ps += __shfl_xor(ps, 1, 16);
      ps += __shfl_xor(ps, 2, 16);
      ps += __shfl_xor(ps, 4, 16);
      ps += __shfl_xor(ps, 8, 16);
      rsum[v] = rsum[v] * corr + ps;
      rmax[v] = mnew;
      o0[v] *= corr;
      o1[v] *= corr;
      const int row = v + hi * 8;
      myP[row * 72 + cl]      = (f16)p0;
      myP[row * 72 + cl + 16] = (f16)p1;
      myP[row * 72 + cl + 32] = (f16)p2;
      myP[row * 72 + cl + 48] = (f16)p3;
    }
    asm volatile("s_wait_dscnt 0" ::: "memory");  // per-wave P RAW through LDS

    // O += P @ V  (two 32-key halves of the tile)
#pragma unroll
    for (int kh = 0; kh < 2; ++kh) {
      v16h aP  = load_a_frag(myP + kh * 32, 0, 72, lane);
      v16h bV0 = load_b_frag(Vs + (size_t)kh * 32 * 40, 40, 0,  lane);
      v16h bV1 = load_b_frag(Vs + (size_t)kh * 32 * 40, 40, 16, lane);
      o0 = wmma_f16(aP, bV0, o0);
      o1 = wmma_f16(aP, bV1, o1);
    }
  }

  // Normalize and scatter into [B,N,256] (head-major columns)
#pragma unroll
  for (int v = 0; v < 8; ++v) {
    const float inv = 1.0f / rsum[v];
    const int n = m0 + v + hi * 8;
    f16* op = oatt + ((size_t)(b * NSEQ + n)) * LATENT + h * DHEAD;
    op[cl]      = (f16)(o0[v] * inv);
    op[16 + cl] = (f16)(o1[v] * inv);
  }
}

// ---------------------------------------------------------------------------
static inline size_t align256(size_t x) { return (x + 255) & ~(size_t)255; }

extern "C" void kernel_launch(void* const* d_in, const int* in_sizes, int n_in,
                              void* d_out, int out_size, void* d_ws, size_t ws_size,
                              hipStream_t stream) {
  (void)in_sizes; (void)n_in; (void)out_size; (void)ws_size;
  const float* A     = (const float*)d_in[0];   // [2,2048,768]
  const float* B     = (const float*)d_in[1];
  const float* Wq_aa = (const float*)d_in[2];   const float* bq_aa = (const float*)d_in[3];
  const float* Wk_aa = (const float*)d_in[4];   const float* bk_aa = (const float*)d_in[5];
  const float* Wv_a  = (const float*)d_in[6];   const float* bv_a  = (const float*)d_in[7];
  const float* Wk_ab = (const float*)d_in[8];   const float* bk_ab = (const float*)d_in[9];
  const float* Wq_bb = (const float*)d_in[10];  const float* bq_bb = (const float*)d_in[11];
  const float* Wo    = (const float*)d_in[12];  const float* bo    = (const float*)d_in[13];

  const int Bsz = 2, M = Bsz * NSEQ;            // 4096
  const int DIM = 768;
  const size_t nX = (size_t)M * DIM;            // A/B elements
  const size_t nW = (size_t)DIM * LATENT;       // per projection weight
  const size_t nWo = (size_t)LATENT * LATENT;
  const size_t nP = (size_t)M * LATENT;         // per projection output

  // Workspace carve-up (f16 buffers)
  char* p = (char*)d_ws;
  size_t off = 0;
  auto grab = [&](size_t bytes) { char* r = p + off; off = align256(off + bytes); return r; };
  f16* hA    = (f16*)grab(nX * 2);
  f16* hB    = (f16*)grab(nX * 2);
  f16* hWqaa = (f16*)grab(nW * 2);
  f16* hWkaa = (f16*)grab(nW * 2);
  f16* hWva  = (f16*)grab(nW * 2);
  f16* hWkab = (f16*)grab(nW * 2);
  f16* hWqbb = (f16*)grab(nW * 2);
  f16* hWo   = (f16*)grab(nWo * 2);
  f16* qsb   = (f16*)grab(nP * 2);   // Q_self  [B,H,N,32]
  f16* kstb  = (f16*)grab(nP * 2);   // K_self  [B,H,32,N]
  f16* qcb   = (f16*)grab(nP * 2);   // Q_cross [B,H,N,32]
  f16* kctb  = (f16*)grab(nP * 2);   // K_cross [B,H,32,N]
  f16* vb    = (f16*)grab(nP * 2);   // V       [B,H,N,32]
  f16* oatt  = (f16*)grab(nP * 2);   // attn out [B,N,256]

  // 1) f32 -> f16 conversions
  auto cvt = [&](const float* src, f16* dst, size_t n) {
    int nn = (int)n;
    cvt_f32_to_f16_kernel<<<(nn + 255) / 256, 256, 0, stream>>>(src, dst, nn);
  };
  cvt(A, hA, nX);        cvt(B, hB, nX);
  cvt(Wq_aa, hWqaa, nW); cvt(Wk_aa, hWkaa, nW); cvt(Wv_a, hWva, nW);
  cvt(Wk_ab, hWkab, nW); cvt(Wq_bb, hWqbb, nW); cvt(Wo, hWo, nWo);

  // 2) Projections (WMMA GEMMs)
  dim3 gp(LATENT / 64, M / 64);   // (4, 64)
  gemm_f16_kernel<0><<<gp, 128, 0, stream>>>(hA, hWqaa, bq_aa, qsb,  M, DIM, LATENT);
  gemm_f16_kernel<1><<<gp, 128, 0, stream>>>(hA, hWkaa, bk_aa, kstb, M, DIM, LATENT);
  gemm_f16_kernel<0><<<gp, 128, 0, stream>>>(hA, hWva,  bv_a,  vb,   M, DIM, LATENT);
  gemm_f16_kernel<0><<<gp, 128, 0, stream>>>(hA, hWkab, bk_ab, qcb,  M, DIM, LATENT);
  gemm_f16_kernel<1><<<gp, 128, 0, stream>>>(hB, hWqbb, bq_bb, kctb, M, DIM, LATENT);

  // 3) Fused dual-score flash attention (LDS-shared K/V tiles)
  dim3 ga(NSEQ / 128, Bsz * HEADS);  // (16, 16)
  attn_flash_kernel<<<ga, 256, 0, stream>>>(qsb, kstb, qcb, kctb, vb, oatt);

  // 4) Output projection -> f32 d_out
  dim3 go(LATENT / 64, M / 64);
  gemm_f16_kernel<2><<<go, 128, 0, stream>>>(oatt, hWo, bo, (float*)d_out,
                                             M, LATENT, LATENT);
}
